// Decoder_47717086658663
// MI455X (gfx1250) — matrix operand order repaired and checked by reading, not compiled, not run on hardware
//
#include <hip/hip_runtime.h>
#include <math.h>

// ---------------- problem constants ----------------
#define Bc   2
#define Sc   1024
#define Ec   1024
#define HQc  16
#define HKVc 4
#define Dc   64
#define WINc 512
#define NEc  8
#define HIDc 4096
#define Mtok (Bc*Sc)          // 2048 tokens
#define EPSc 1.1920929e-07f

// ---------------- WMMA plumbing ----------------
typedef __attribute__((ext_vector_type(16))) __bf16        bf16x16;
typedef __attribute__((ext_vector_type(8)))  float         f32x8;
typedef __attribute__((ext_vector_type(4)))  unsigned int  u32x4;
typedef __attribute__((ext_vector_type(4)))  float         f32x4;

__device__ __forceinline__ f32x8 wmma_bf16(bf16x16 a, bf16x16 b, f32x8 c) {
  // D = A(16x32 bf16) * B(32x16 bf16) + C(16x16 f32)
  return __builtin_amdgcn_wmma_f32_16x16x32_bf16(false, a, false, b,
                                                 (short)0, c, false, false);
}

// A-fragment (16x32 bf16) from row-major [16][stride] buffer.
// Contiguous k order used here (B matches): lane<16 -> M=lane, K=0..15; lane>=16 -> M=lane-16, K=16..31
__device__ __forceinline__ bf16x16 load_a_frag(const __bf16* base, int stride, int lane) {
  int row  = lane & 15;
  int koff = (lane >> 4) * 16;
  const __bf16* p = base + row * stride + koff;
  union { bf16x16 v; u32x4 q[2]; } u;
  u.q[0] = *(const u32x4*)(p);
  u.q[1] = *(const u32x4*)(p + 8);
  return u.v;
}

// B-fragment (32x16 bf16) from W^T tile [n][k-contig]:
// lane<16 -> N=lane, K=0..15 ; lane>=16 -> N=lane-16, K=16..31
__device__ __forceinline__ bf16x16 load_b_frag(const __bf16* base, int stride, int lane) {
  int n  = lane & 15;
  int k0 = (lane >> 4) * 16;
  const __bf16* p = base + n * stride + k0;
  union { bf16x16 v; u32x4 q[2]; } u;
  u.q[0] = *(const u32x4*)(p);
  u.q[1] = *(const u32x4*)(p + 8);
  return u.v;
}

// ---------------- RMSNorm (fp32 in -> bf16 out) ----------------
__global__ void rmsnorm_kernel(const float* __restrict__ x,
                               const float* __restrict__ w,
                               __bf16* __restrict__ out) {
  int row = blockIdx.x;
  const float* xr = x + (size_t)row * Ec;
  float ss = 0.f;
  for (int i = threadIdx.x; i < Ec; i += blockDim.x) { float v = xr[i]; ss += v * v; }
  for (int m = 16; m; m >>= 1) ss += __shfl_xor(ss, m, 32);
  __shared__ float red[8];
  int wid = threadIdx.x >> 5;
  if ((threadIdx.x & 31) == 0) red[wid] = ss;
  __syncthreads();
  if (threadIdx.x == 0) {
    float v = 0.f;
    for (int j = 0; j < 8; ++j) v += red[j];
    red[0] = v;
  }
  __syncthreads();
  float inv = rsqrtf(red[0] / (float)Ec + EPSc);
  for (int i = threadIdx.x; i < Ec; i += blockDim.x)
    out[(size_t)row * Ec + i] = (__bf16)(xr[i] * inv * w[i]);
}

// ============ shared tile-loader helpers (128x32 A, 64x32 W^T, 256 threads) =====
#define ASTR 40   // 32 + 8 pad, keeps 16B alignment
#define TMr  128

// Load A tile rows [mbase, mbase+128) x K-chunk 32 into As (row-major).
__device__ __forceinline__ void load_a_tile(const __bf16* __restrict__ A, __bf16* As,
                                            int gmRow, int valid, size_t rowStride,
                                            int k0, int tid, int K) {
  int row = tid >> 1, half = (tid & 1) * 16;
  u32x4* dst = (u32x4*)&As[row * ASTR + half];
  if (valid) {
    const __bf16* src = A + (size_t)gmRow * rowStride + k0 + half;
    dst[0] = *(const u32x4*)src;
    dst[1] = *(const u32x4*)(src + 8);
    // prefetch next K chunk of this row
    if (k0 + 32 < K) __builtin_prefetch((const void*)(src + 32), 0, 1);
  } else {
    u32x4 zz = {0,0,0,0}; dst[0] = zz; dst[1] = zz;
  }
}

// Load W tile fp32 [K,N] chunk (32 x 64) -> bf16 transposed Ws[n][k].
// Coalesced: 16 lanes sweep one K-row with float4 loads.
__device__ __forceinline__ void load_w_tile(const float* __restrict__ W, __bf16* Ws,
                                            int N, int nbase, int k0, int tid, int K) {
  int n4   = (tid & 15) * 4;
  int krow = tid >> 4;           // 0..15
#pragma unroll
  for (int j = 0; j < 2; ++j) {
    int k = krow + j * 16;
    const float* src = &W[(size_t)(k0 + k) * N + nbase + n4];
    f32x4 w4 = *(const f32x4*)src;
    if (k0 + 32 < K) __builtin_prefetch((const void*)(src + (size_t)32 * N), 0, 1);
#pragma unroll
    for (int i = 0; i < 4; ++i) Ws[(n4 + i) * ASTR + k] = (__bf16)w4[i];
  }
}

// ---------------- generic WMMA GEMM:  C[M,N] = A_bf16[M,K] * W_f32[K,N] + bias ----
// mode 0: out idx = m*N+n (+ optional residual);  mode 1: head remap
__global__ __launch_bounds__(256) void wmma_gemm_kernel(
    const __bf16* __restrict__ A, const float* __restrict__ W,
    const float* __restrict__ bias, const float* __restrict__ resid,
    float* __restrict__ outF, __bf16* __restrict__ outB,
    int M, int N, int K, int mode, int H) {
  __shared__ __bf16 As[TMr * ASTR];
  __shared__ __bf16 Ws_[64 * ASTR];
  int mbase = blockIdx.y * TMr, nbase = blockIdx.x * 64;
  int tid = threadIdx.x, lane = tid & 31, wave = tid >> 5;
  f32x8 z8 = {0,0,0,0,0,0,0,0};
  f32x8 acc[4] = {z8, z8, z8, z8};
  for (int k0 = 0; k0 < K; k0 += 32) {
    __syncthreads();
    int arow = mbase + (tid >> 1);
    load_a_tile(A, As, arow, arow < M, K, k0, tid, K);
    load_w_tile(W, Ws_, N, nbase, k0, tid, K);
    __syncthreads();
    bf16x16 a = load_a_frag(&As[wave * 16 * ASTR], ASTR, lane);
#pragma unroll
    for (int f = 0; f < 4; ++f) {
      bf16x16 b = load_b_frag(&Ws_[f * 16 * ASTR], ASTR, lane);
      acc[f] = wmma_bf16(a, b, acc[f]);
    }
  }
  int mrow = mbase + wave * 16 + 8 * (lane >> 4);
#pragma unroll
  for (int f = 0; f < 4; ++f) {
    int n = nbase + f * 16 + (lane & 15);
    if (n >= N) continue;
    float bv = bias ? bias[n] : 0.f;
#pragma unroll
    for (int r = 0; r < 8; ++r) {
      int m = mrow + r;
      if (m >= M) continue;
      float v = acc[f][r] + bv;
      size_t oidx;
      if (mode == 0) {
        oidx = (size_t)m * N + n;
        if (resid) v += resid[oidx];
      } else {
        int b = m / Sc, s = m % Sc, hh = n >> 6, d = n & 63;
        oidx = (((size_t)b * H + hh) * Sc + s) * 64 + d;
      }
      if (outF) outF[oidx] = v;
      if (outB) outB[oidx] = (__bf16)v;
    }
  }
}

// ---------------- RoPE + cast (+score prescale for q) ----------------
__global__ void rope_cast_kernel(const float* __restrict__ in, __bf16* __restrict__ out,
                                 int total, float scale) {
  int idx = blockIdx.x * blockDim.x + threadIdx.x;
  if (idx >= total) return;
  int d = idx & 31;
  int s = (idx >> 5) % Sc;
  int bh = idx / (32 * Sc);
  float inv = __powf(10000.f, -(float)d / 32.f);
  float ang = (float)s * inv;
  float c = __cosf(ang), sn = __sinf(ang);
  size_t base = ((size_t)bh * Sc + s) * 64;
  float x1 = in[base + d], x2 = in[base + 32 + d];
  out[base + d]      = (__bf16)((x1 * c - x2 * sn) * scale);
  out[base + 32 + d] = (__bf16)((x1 * sn + x2 * c) * scale);
}

// ---------------- sliding-window flash attention (1 wave / 16-query tile) -------
#define VSTR 40   // key stride in transposed V tile
__global__ __launch_bounds__(32) void attn_kernel(
    const __bf16* __restrict__ qb, const __bf16* __restrict__ kb,
    const __bf16* __restrict__ vb, __bf16* __restrict__ ao) {
  __shared__ __bf16 Pl[16 * 32];
  __shared__ __bf16 Vt[64 * VSTR];   // transposed: [d][key]
  int lane = threadIdx.x;
  int qtile = blockIdx.x, bh = blockIdx.y;
  int b = bh / HQc, h = bh % HQc, kvh = h / (HQc / HKVc);
  int q0 = qtile * 16;
  const __bf16* qp = qb + ((size_t)bh * Sc + q0) * 64;
  const __bf16* kp = kb + ((size_t)(b * HKVc + kvh) * Sc) * 64;
  const __bf16* vp = vb + ((size_t)(b * HKVc + kvh) * Sc) * 64;

  bf16x16 qa[2];
  {
    int row = lane & 15, koff = (lane >> 4) * 16;
#pragma unroll
    for (int c = 0; c < 2; ++c) {
      const __bf16* p = qp + row * 64 + c * 32 + koff;
      union { bf16x16 v; u32x4 q[2]; } u;
      u.q[0] = *(const u32x4*)p; u.q[1] = *(const u32x4*)(p + 8);
      qa[c] = u.v;
    }
  }
  f32x8 z8 = {0,0,0,0,0,0,0,0};
  f32x8 acc[4] = {z8, z8, z8, z8};
  float mrow[8], lrow[8];
#pragma unroll
  for (int r = 0; r < 8; ++r) { mrow[r] = -1e30f; lrow[r] = 0.f; }

  int kb0 = q0 - WINc; if (kb0 < 0) kb0 = 0; kb0 &= ~31;
  int kend = q0 + 16;
  int colbase = lane & 15, rbase = q0 + 8 * (lane >> 4);

  for (int kbs = kb0; kbs < kend; kbs += 32) {
    // ---- stage V transposed early (latency-tolerant scalar stores) ----
    {
      int key = kbs + lane; if (key > Sc - 1) key = Sc - 1;
      const __bf16* src = vp + (size_t)key * 64;
      union { u32x4 q; __bf16 e[8]; } u;
#pragma unroll
      for (int j = 0; j < 8; ++j) {
        u.q = *(const u32x4*)(src + j * 8);
#pragma unroll
        for (int i = 0; i < 8; ++i) Vt[(j * 8 + i) * VSTR + lane] = u.e[i];
      }
      if (kbs + 32 < kend) __builtin_prefetch((const void*)(src + 32 * 64), 0, 1);
    }
    // ---- scores: two 16x16 tiles over 32 keys, K-dim = D=64 (2 chunks) ----
    f32x8 sc[2];
#pragma unroll
    for (int hh = 0; hh < 2; ++hh) {
      f32x8 s = z8;
      int key = kbs + hh * 16 + (lane & 15); if (key > Sc - 1) key = Sc - 1;
#pragma unroll
      for (int c = 0; c < 2; ++c) {
        int d0 = c * 32 + (lane >> 4) * 16;
        const __bf16* p = kp + (size_t)key * 64 + d0;
        union { bf16x16 v; u32x4 q[2]; } u;
        u.q[0] = *(const u32x4*)p; u.q[1] = *(const u32x4*)(p + 8);
        s = wmma_bf16(qa[c], u.v, s);
      }
      sc[hh] = s;
    }
    // ---- mask + online softmax (row stats via width-16 butterflies) ----
    float pnew[2][8];
#pragma unroll
    for (int r = 0; r < 8; ++r) {
      int row = rbase + r;
      float mx = -1e30f, sv[2];
#pragma unroll
      for (int hh = 0; hh < 2; ++hh) {
        int col = kbs + hh * 16 + colbase;
        float v = sc[hh][r];
        bool ok = (col <= row) && (col >= row - WINc);
        v = ok ? v : -1e30f;
        sv[hh] = v; mx = fmaxf(mx, v);
      }
      for (int msk = 1; msk < 16; msk <<= 1) mx = fmaxf(mx, __shfl_xor(mx, msk, 16));
      float nm = fmaxf(mrow[r], mx);
      float scl = __expf(mrow[r] - nm);
      float psum = 0.f;
#pragma unroll
      for (int hh = 0; hh < 2; ++hh) { float p = __expf(sv[hh] - nm); pnew[hh][r] = p; psum += p; }
      for (int msk = 1; msk < 16; msk <<= 1) psum += __shfl_xor(psum, msk, 16);
      lrow[r] = lrow[r] * scl + psum;
      mrow[r] = nm;
#pragma unroll
      for (int f = 0; f < 4; ++f) acc[f][r] *= scl;
    }
    // ---- stage P (C-layout -> row-major LDS -> A-layout) ----
#pragma unroll
    for (int hh = 0; hh < 2; ++hh)
#pragma unroll
      for (int r = 0; r < 8; ++r)
        Pl[(r + 8 * (lane >> 4)) * 32 + hh * 16 + colbase] = (__bf16)pnew[hh][r];
    __syncthreads();
    bf16x16 pa = load_a_frag(Pl, 32, lane);
#pragma unroll
    for (int f = 0; f < 4; ++f) {
      // B-frag from transposed V: N=d, K=keys contiguous -> two b128 LDS reads
      int d = f * 16 + (lane & 15);
      const __bf16* p = &Vt[d * VSTR + (lane >> 4) * 16];
      union { bf16x16 v; u32x4 q[2]; } u;
      u.q[0] = *(const u32x4*)p; u.q[1] = *(const u32x4*)(p + 8);
      acc[f] = wmma_bf16(pa, u.v, acc[f]);
    }
    __syncthreads();
  }
  // ---- epilogue: O /= l, write bf16 [token, HQ*64] ----
#pragma unroll
  for (int r = 0; r < 8; ++r) {
    int row = rbase + r;
    float inv = 1.f / lrow[r];
    int token = b * Sc + row;
#pragma unroll
    for (int f = 0; f < 4; ++f) {
      int col = h * 64 + f * 16 + colbase;
      ao[(size_t)token * (HQc * 64) + col] = (__bf16)(acc[f][r] * inv);
    }
  }
}

// ---------------- router: logits, top-2 softmax, counts ----------------
__global__ void router_kernel(const __bf16* __restrict__ t, const float* __restrict__ Wr,
                              const float* __restrict__ br, int* __restrict__ eid,
                              float* __restrict__ pw, int* __restrict__ counts) {
  int tok = blockIdx.x;
  __shared__ float lg[NEc];
  __shared__ float red[8];
  const __bf16* tr = t + (size_t)tok * Ec;
  for (int e = 0; e < NEc; ++e) {
    float s = 0.f;
    for (int i = threadIdx.x; i < Ec; i += blockDim.x)
      s += (float)tr[i] * Wr[(size_t)i * NEc + e];
    for (int m = 16; m; m >>= 1) s += __shfl_xor(s, m, 32);
    if ((threadIdx.x & 31) == 0) red[threadIdx.x >> 5] = s;
    __syncthreads();
    if (threadIdx.x == 0) {
      float v = 0.f;
      for (int j = 0; j < 8; ++j) v += red[j];
      lg[e] = v + br[e];
    }
    __syncthreads();
  }
  if (threadIdx.x == 0) {
    int i0 = 0;
    for (int e = 1; e < NEc; ++e) if (lg[e] > lg[i0]) i0 = e;
    int i1 = -1;
    for (int e = 0; e < NEc; ++e) { if (e == i0) continue; if (i1 < 0 || lg[e] > lg[i1]) i1 = e; }
    float mx = fmaxf(lg[i0], lg[i1]);
    float e0 = __expf(lg[i0] - mx), e1 = __expf(lg[i1] - mx);
    float zz = e0 + e1;
    eid[tok * 2 + 0] = i0; eid[tok * 2 + 1] = i1;
    pw[tok * 2 + 0] = e0 / zz; pw[tok * 2 + 1] = e1 / zz;
    atomicAdd(&counts[i0], 1); atomicAdd(&counts[i1], 1);
  }
}

__global__ void offsets_kernel(const int* __restrict__ counts, int* __restrict__ offs) {
  if (threadIdx.x == 0 && blockIdx.x == 0) {
    int a = 0;
    for (int e = 0; e < NEc; ++e) { offs[e] = a; a += counts[e]; }
  }
}

__global__ void scatter_kernel(const int* __restrict__ eid, const int* __restrict__ offs,
                               int* __restrict__ fill, int* __restrict__ slot_tok,
                               int* __restrict__ tok_slot) {
  int tok = blockIdx.x * blockDim.x + threadIdx.x;
  if (tok >= Mtok) return;
  for (int j = 0; j < 2; ++j) {
    int e = eid[tok * 2 + j];
    int pos = atomicAdd(&fill[e], 1);
    int slot = offs[e] + pos;
    slot_tok[slot] = tok;
    tok_slot[tok * 2 + j] = slot;
  }
}

// ---------------- MoE grouped GEMM 1: h1 = silu(t*Wg+gb) * (t*Wu+ub) -> bf16 ----
__global__ __launch_bounds__(256) void moe_gemm1_kernel(
    const __bf16* __restrict__ Tb, const float* __restrict__ gate_w,
    const float* __restrict__ gate_b, const float* __restrict__ up_w,
    const float* __restrict__ up_b, const int* __restrict__ counts,
    const int* __restrict__ offs, const int* __restrict__ slot_tok,
    __bf16* __restrict__ h1) {
  int e = blockIdx.z;
  int rows = counts[e];
  int mbase = blockIdx.y * TMr;
  if (mbase >= rows) return;
  int nbase = blockIdx.x * 64;
  const float* Wg = gate_w + (size_t)e * Ec * HIDc;
  const float* Wu = up_w   + (size_t)e * Ec * HIDc;
  __shared__ __bf16 As[TMr * ASTR];
  __shared__ __bf16 Gs[64 * ASTR];
  __shared__ __bf16 Us[64 * ASTR];
  int tid = threadIdx.x, lane = tid & 31, wave = tid >> 5;
  f32x8 z8 = {0,0,0,0,0,0,0,0};
  f32x8 ag[4] = {z8, z8, z8, z8}, au[4] = {z8, z8, z8, z8};
  int m_ld = mbase + (tid >> 1);
  int tokr = (m_ld < rows) ? slot_tok[offs[e] + m_ld] : 0;
  for (int k0 = 0; k0 < Ec; k0 += 32) {
    __syncthreads();
    load_a_tile(Tb, As, tokr, m_ld < rows, Ec, k0, tid, Ec);
    load_w_tile(Wg, Gs, HIDc, nbase, k0, tid, Ec);
    load_w_tile(Wu, Us, HIDc, nbase, k0, tid, Ec);
    __syncthreads();
    bf16x16 a = load_a_frag(&As[wave * 16 * ASTR], ASTR, lane);
#pragma unroll
    for (int f = 0; f < 4; ++f) {
      bf16x16 bg = load_b_frag(&Gs[f * 16 * ASTR], ASTR, lane);
      ag[f] = wmma_bf16(a, bg, ag[f]);
      bf16x16 bu = load_b_frag(&Us[f * 16 * ASTR], ASTR, lane);
      au[f] = wmma_bf16(a, bu, au[f]);
    }
  }
  int mloc = wave * 16 + 8 * (lane >> 4);
#pragma unroll
  for (int f = 0; f < 4; ++f) {
    int n = nbase + f * 16 + (lane & 15);
    float gb = gate_b[(size_t)e * HIDc + n];
    float ub = up_b[(size_t)e * HIDc + n];
#pragma unroll
    for (int r = 0; r < 8; ++r) {
      int m = mbase + mloc + r;
      if (m >= rows) continue;
      float g = ag[f][r] + gb;
      float u = au[f][r] + ub;
      float sg = g / (1.f + __expf(-g));
      int slot = offs[e] + m;
      h1[(size_t)slot * HIDc + n] = (__bf16)(u * sg);
    }
  }
}

// ---------------- MoE grouped GEMM 2: y = h1 * Wd + db (f32 per slot) -----------
__global__ __launch_bounds__(256) void moe_gemm2_kernel(
    const __bf16* __restrict__ h1, const float* __restrict__ down_w,
    const float* __restrict__ down_b, const int* __restrict__ counts,
    const int* __restrict__ offs, float* __restrict__ y) {
  int e = blockIdx.z;
  int rows = counts[e];
  int mbase = blockIdx.y * TMr;
  if (mbase >= rows) return;
  int nbase = blockIdx.x * 64;
  const float* Wd = down_w + (size_t)e * HIDc * Ec;
  const __bf16* Ab = h1 + (size_t)offs[e] * HIDc;
  __shared__ __bf16 As[TMr * ASTR];
  __shared__ __bf16 Ws_[64 * ASTR];
  int tid = threadIdx.x, lane = tid & 31, wave = tid >> 5;
  f32x8 z8 = {0,0,0,0,0,0,0,0};
  f32x8 acc[4] = {z8, z8, z8, z8};
  int m_ld = mbase + (tid >> 1);
  for (int k0 = 0; k0 < HIDc; k0 += 32) {
    __syncthreads();
    load_a_tile(Ab, As, m_ld, m_ld < rows, HIDc, k0, tid, HIDc);
    load_w_tile(Wd, Ws_, Ec, nbase, k0, tid, HIDc);
    __syncthreads();
    bf16x16 a = load_a_frag(&As[wave * 16 * ASTR], ASTR, lane);
#pragma unroll
    for (int f = 0; f < 4; ++f) {
      bf16x16 b = load_b_frag(&Ws_[f * 16 * ASTR], ASTR, lane);
      acc[f] = wmma_bf16(a, b, acc[f]);
    }
  }
  int mloc = wave * 16 + 8 * (lane >> 4);
#pragma unroll
  for (int f = 0; f < 4; ++f) {
    int n = nbase + f * 16 + (lane & 15);
    float db = down_b[(size_t)e * Ec + n];
#pragma unroll
    for (int r = 0; r < 8; ++r) {
      int m = mbase + mloc + r;
      if (m >= rows) continue;
      y[(size_t)(offs[e] + m) * Ec + n] = acc[f][r] + db;
    }
  }
}

// ---------------- final combine: out = x2 + p0*y[s0] + p1*y[s1] ----------------
__global__ void combine_kernel(const float* __restrict__ x2, const float* __restrict__ y,
                               const int* __restrict__ tok_slot, const float* __restrict__ pw,
                               float* __restrict__ out) {
  int tok = blockIdx.x;
  int s0 = tok_slot[tok * 2], s1 = tok_slot[tok * 2 + 1];
  float p0 = pw[tok * 2], p1 = pw[tok * 2 + 1];
  for (int i = threadIdx.x; i < Ec; i += blockDim.x)
    out[(size_t)tok * Ec + i] = x2[(size_t)tok * Ec + i]
                              + p0 * y[(size_t)s0 * Ec + i]
                              + p1 * y[(size_t)s1 * Ec + i];
}

// ---------------- host side ----------------
static inline size_t alignup(size_t x) { return (x + 255) & ~(size_t)255; }

extern "C" void kernel_launch(void* const* d_in, const int* in_sizes, int n_in,
                              void* d_out, int out_size, void* d_ws, size_t ws_size,
                              hipStream_t stream) {
  (void)in_sizes; (void)n_in; (void)out_size; (void)ws_size;
  const float* x       = (const float*)d_in[0];
  const float* norm1_w = (const float*)d_in[1];
  const float* norm2_w = (const float*)d_in[2];
  const float* Wq = (const float*)d_in[3];  const float* bq = (const float*)d_in[4];
  const float* Wk = (const float*)d_in[5];  const float* bk = (const float*)d_in[6];
  const float* Wv = (const float*)d_in[7];  const float* bv = (const float*)d_in[8];
  const float* Wo = (const float*)d_in[9];  const float* bo = (const float*)d_in[10];
  const float* Wr = (const float*)d_in[11]; const float* br = (const float*)d_in[12];
  const float* gate_w = (const float*)d_in[13]; const float* gate_b = (const float*)d_in[14];
  const float* up_w   = (const float*)d_in[15]; const float* up_b   = (const float*)d_in[16];
  const float* down_w = (const float*)d_in[17]; const float* down_b = (const float*)d_in[18];
  float* out = (float*)d_out;

  char* ws = (char*)d_ws;
  size_t off = 0;
  __bf16* hb = (__bf16*)(ws + off); off = alignup(off + (size_t)Mtok * Ec * 2);
  float*  qf = (float*) (ws + off); off = alignup(off + (size_t)Mtok * Ec * 4);
  float*  kf = (float*) (ws + off); off = alignup(off + (size_t)Mtok * HKVc * Dc * 4);
  __bf16* vbb= (__bf16*)(ws + off); off = alignup(off + (size_t)Mtok * HKVc * Dc * 2);
  __bf16* qbb= (__bf16*)(ws + off); off = alignup(off + (size_t)Mtok * Ec * 2);
  __bf16* kbb= (__bf16*)(ws + off); off = alignup(off + (size_t)Mtok * HKVc * Dc * 2);
  __bf16* ao = (__bf16*)(ws + off); off = alignup(off + (size_t)Mtok * Ec * 2);
  float*  x2 = (float*) (ws + off); off = alignup(off + (size_t)Mtok * Ec * 4);
  __bf16* tb = (__bf16*)(ws + off); off = alignup(off + (size_t)Mtok * Ec * 2);
  int* counts = (int*)(ws + off);           // 8 ints
  int* fill   = counts + 8;                 // 8 ints
  int* offs   = counts + 16;                // 8 ints
  off = alignup(off + 256);
  int*   eid      = (int*)  (ws + off); off = alignup(off + (size_t)Mtok * 2 * 4);
  float* pw       = (float*)(ws + off); off = alignup(off + (size_t)Mtok * 2 * 4);
  int*   slot_tok = (int*)  (ws + off); off = alignup(off + (size_t)Mtok * 2 * 4);
  int*   tok_slot = (int*)  (ws + off); off = alignup(off + (size_t)Mtok * 2 * 4);
  __bf16* h1 = (__bf16*)(ws + off); off = alignup(off + (size_t)Mtok * 2 * HIDc * 2);
  float*  y  = (float*) (ws + off); off = alignup(off + (size_t)Mtok * 2 * Ec * 4);

  // 1) RMSNorm -> bf16 h
  rmsnorm_kernel<<<Mtok, 256, 0, stream>>>(x, norm1_w, hb);

  // 2) QKV projections (WMMA), head-remapped outputs
  wmma_gemm_kernel<<<dim3(16, Mtok / TMr), 256, 0, stream>>>(hb, Wq, bq, nullptr, qf, nullptr,
                                                             Mtok, HQc * Dc, Ec, 1, HQc);
  wmma_gemm_kernel<<<dim3(4, Mtok / TMr), 256, 0, stream>>>(hb, Wk, bk, nullptr, kf, nullptr,
                                                            Mtok, HKVc * Dc, Ec, 1, HKVc);
  wmma_gemm_kernel<<<dim3(4, Mtok / TMr), 256, 0, stream>>>(hb, Wv, bv, nullptr, nullptr, vbb,
                                                            Mtok, HKVc * Dc, Ec, 1, HKVc);

  // 3) RoPE + bf16 cast (q carries 1/sqrt(D) score scale)
  {
    int totq = Bc * HQc * Sc * 32;
    rope_cast_kernel<<<(totq + 255) / 256, 256, 0, stream>>>(qf, qbb, totq, 0.125f);
    int totk = Bc * HKVc * Sc * 32;
    rope_cast_kernel<<<(totk + 255) / 256, 256, 0, stream>>>(kf, kbb, totk, 1.0f);
  }

  // 4) sliding-window flash attention
  attn_kernel<<<dim3(Sc / 16, Bc * HQc), 32, 0, stream>>>(qbb, kbb, vbb, ao);

  // 5) o-proj + bias + residual
  wmma_gemm_kernel<<<dim3(16, Mtok / TMr), 256, 0, stream>>>(ao, Wo, bo, x, x2, nullptr,
                                                             Mtok, Ec, Ec, 0, 0);

  // 6) RMSNorm2 -> bf16 t
  rmsnorm_kernel<<<Mtok, 256, 0, stream>>>(x2, norm2_w, tb);

  // 7) routing: logits -> top2 -> scatter into per-expert contiguous slots
  hipMemsetAsync(counts, 0, 64, stream);  // counts + fill
  router_kernel<<<Mtok, 256, 0, stream>>>(tb, Wr, br, eid, pw, counts);
  offsets_kernel<<<1, 32, 0, stream>>>(counts, offs);
  scatter_kernel<<<(Mtok + 255) / 256, 256, 0, stream>>>(eid, offs, fill, slot_tok, tok_slot);

  // 8) grouped expert FFN (early-exit on empty tiles)
  moe_gemm1_kernel<<<dim3(HIDc / 64, 16, NEc), 256, 0, stream>>>(
      tb, gate_w, gate_b, up_w, up_b, counts, offs, slot_tok, h1);
  moe_gemm2_kernel<<<dim3(Ec / 64, 16, NEc), 256, 0, stream>>>(
      h1, down_w, down_b, counts, offs, y);

  // 9) combine + residual
  combine_kernel<<<Mtok, 256, 0, stream>>>(x2, y, tok_slot, pw, out);
}
